// WindowAttention_90142773608533
// MI455X (gfx1250) — matrix operand order repaired and checked by reading, not compiled, not run on hardware
//
#include <hip/hip_runtime.h>
#include <hip/hip_bf16.h>

typedef __attribute__((ext_vector_type(16))) __bf16 v16bf;
typedef __attribute__((ext_vector_type(8)))  float  v8f;

#define WSZ 7
#define NHEAD 4
#define CH 128
#define NTOK 49
#define NPAD 64
#define NWINDOWS 4096
#define NWIMG 64
#define C3 384
#define DH 32

// LDS pitches in elements (padded: row stride mod 64 banks = 4; 36 for the 72-bf16 pitches)
#define QKV_PITCH 264   // 256+8 bf16 (row = 528B, 16B aligned)   q|k only
#define X_PITCH   136   // 128+8 bf16 (row = 272B)
#define VT_PITCH  72    // 64+8 bf16  (row = 144B)
#define AT_PITCH  68    // 64+4 f32   (row = 272B)
#define AB_PITCH  72    // 64+8 bf16

// LDS byte offsets
#define OFF_QKV  0
#define OFF_VT   33792                    // 64*264*2
#define OFF_ATT  52224                    // +128*72*2
#define OFF_POOL 121856                   // +4*64*68*4
#define OFF_CTX  (OFF_POOL + 36864)       // attnb = 4*64*72*2
#define SMEM_BYTES (OFF_CTX + 17408)      // +64*136*2 = 176128

// d_ws byte offsets
#define WS_WQKVT 0
#define WS_PROJT (C3*CH*2)                 // 98304
#define WS_RPB   (WS_PROJT + CH*CH*2)      // 131072

// Load a 16-element bf16 fragment as two 16B chunks (matches CDNA5
// 16-bit A/B VGPR layouts: elements 0-7 then 8-15).
__device__ __forceinline__ v16bf frag2(const __bf16* p0, const __bf16* p1) {
  union { v16bf v; uint4 q[2]; } u;
  u.q[0] = *(const uint4*)p0;
  u.q[1] = *(const uint4*)p1;
  return u.v;
}

__global__ __launch_bounds__(256) void swin_prep(
    const float* __restrict__ qkv_w, const float* __restrict__ proj_w,
    const float* __restrict__ bias_table,
    __bf16* __restrict__ wqkvT, __bf16* __restrict__ projT,
    float* __restrict__ rpb)
{
  int i = blockIdx.x * blockDim.x + threadIdx.x;
  if (i < C3*CH) {                         // wqkvT[n][k] = qkv_w[k][n]
    int n = i / CH, k = i % CH;
    wqkvT[i] = (__bf16)qkv_w[k*C3 + n];
  } else if (i < C3*CH + CH*CH) {          // projT[n][k] = proj_w[k][n]
    int j = i - C3*CH;
    int n = j / CH, k = j % CH;
    projT[j] = (__bf16)proj_w[k*CH + n];
  } else if (i < C3*CH + CH*CH + NHEAD*NTOK*NTOK) {
    int j = i - C3*CH - CH*CH;
    int h = j / (NTOK*NTOK), p = j % (NTOK*NTOK);
    int a = p / NTOK, b = p % NTOK;
    int ra = a / WSZ, ca = a % WSZ, rb = b / WSZ, cb = b % WSZ;
    int idx = (ra - rb + WSZ - 1) * (2*WSZ - 1) + (ca - cb + WSZ - 1);
    rpb[j] = bias_table[idx*NHEAD + h];
  }
}

__global__ __launch_bounds__(256) void swin_win_attn(
    const float* __restrict__ x, const float* __restrict__ qkv_b,
    const float* __restrict__ proj_b, const float* __restrict__ mask,
    const __bf16* __restrict__ wqkvT, const __bf16* __restrict__ projT,
    const float* __restrict__ rpb, float* __restrict__ out)
{
  extern __shared__ __align__(16) char smem[];
  __bf16* s_qkv = (__bf16*)(smem + OFF_QKV);   // [64][264] bf16: q(scaled)|k
  __bf16* s_vT  = (__bf16*)(smem + OFF_VT);    // [128][72]  bf16: v transposed
  float*  s_att = (float*) (smem + OFF_ATT);   // [4][64][68] f32 scores
  __bf16* s_x   = (__bf16*)(smem + OFF_POOL);  // [64][136] bf16 (phases 1-2)
  __bf16* s_ab  = (__bf16*)(smem + OFF_POOL);  // [4][64][72] bf16 softmax (4-5)
  __bf16* s_ctx = (__bf16*)(smem + OFF_CTX);   // [64][136] bf16 context (5-6)

  const int w    = blockIdx.x;
  const int tid  = threadIdx.x;
  // Scalarize the wave id so tile indices/branches become SALU.
  const int wave = __builtin_amdgcn_readfirstlane(tid >> 5);
  const int lane = tid & 31;
  const int hl   = lane >> 4, l16 = lane & 15;
  const float scale = 0.17677669529663687f;    // 1/sqrt(32)

  // ---- Phase 1: x -> bf16 LDS, vectorized (pad rows 49..63 with zero) ----
  const float4* xw4 = (const float4*)(x + (size_t)w * NTOK * CH);
  for (int i = tid; i < NPAD*(CH/4); i += 256) {
    int r = i >> 5, c4 = i & 31;
    float4 v = make_float4(0.f, 0.f, 0.f, 0.f);
    if (r < NTOK) v = xw4[r*(CH/4) + c4];
    union { __bf16 h[4]; uint2 q; } u;
    u.h[0] = (__bf16)v.x; u.h[1] = (__bf16)v.y;
    u.h[2] = (__bf16)v.z; u.h[3] = (__bf16)v.w;
    *(uint2*)(s_x + r*X_PITCH + c4*4) = u.q;
  }
  __syncthreads();

  // ---- Phase 2: qkv = x @ Wqkv + b (q pre-scaled); v goes only to vT ----
  for (int t = wave; t < 96; t += 8) {          // 4 M-tiles x 24 N-tiles
    int nt = t % 24, mt = t / 24;               // SGPRs
    int n = nt*16 + l16;
    v8f acc = {};
    const __bf16* arow = s_x + (mt*16 + l16) * X_PITCH;
    const __bf16* brow = wqkvT + (size_t)n * CH;
    #pragma unroll
    for (int ks = 0; ks < 4; ++ks) {
      v16bf a = frag2(arow + ks*32 + hl*8, arow + ks*32 + 16 + hl*8);
      v16bf b = frag2(brow + ks*32 + hl*8, brow + ks*32 + 16 + hl*8);
      acc = __builtin_amdgcn_wmma_f32_16x16x32_bf16(false, a, false, b,
                                                    (short)0, acc, false, false);
    }
    float bias = qkv_b[n];
    int mbase = mt*16 + 8*hl;
    if (nt < 8) {                               // q section: scale, row-major
      __bf16* crow = s_qkv + n;
      #pragma unroll
      for (int r = 0; r < 8; ++r)
        crow[(mbase + r)*QKV_PITCH] = (__bf16)((acc[r] + bias) * scale);
    } else if (nt < 16) {                       // k section, row-major
      __bf16* crow = s_qkv + n;
      #pragma unroll
      for (int r = 0; r < 8; ++r)
        crow[(mbase + r)*QKV_PITCH] = (__bf16)(acc[r] + bias);
    } else {                                    // v section: transposed only
      union { __bf16 h[8]; uint4 q; } u;
      #pragma unroll
      for (int r = 0; r < 8; ++r)
        u.h[r] = (__bf16)(acc[r] + bias);
      *(uint4*)(s_vT + (n - 2*CH)*VT_PITCH + mbase) = u.q;  // 8 contiguous
    }
  }
  __syncthreads();

  // ---- Phase 3: scores = qs @ k^T + rpb + mask (f32 in LDS) ----
  const float* maskw = mask + (size_t)(w & (NWIMG-1)) * NTOK * NTOK;
  for (int t = wave; t < 64; t += 8) {          // 4 heads x 4x4 tiles
    int h = t >> 4, mt = (t >> 2) & 3, nt2 = t & 3;
    const __bf16* arow = s_qkv + (mt*16  + l16) * QKV_PITCH + h*DH;        // q
    const __bf16* brow = s_qkv + (nt2*16 + l16) * QKV_PITCH + CH + h*DH;   // k
    v16bf a = frag2(arow + hl*8, arow + 16 + hl*8);
    v16bf b = frag2(brow + hl*8, brow + 16 + hl*8);
    v8f acc = {};
    acc = __builtin_amdgcn_wmma_f32_16x16x32_bf16(false, a, false, b,
                                                  (short)0, acc, false, false);
    int m = nt2*16 + l16;
    #pragma unroll
    for (int r = 0; r < 8; ++r) {
      int n = mt*16 + r + 8*hl;
      float v = (n < NTOK && m < NTOK)
                  ? acc[r] + rpb[(h*NTOK + n)*NTOK + m] + maskw[n*NTOK + m]
                  : -1e30f;
      s_att[(h*NPAD + n)*AT_PITCH + m] = v;
    }
  }
  __syncthreads();

  // ---- Phase 4: row softmax, packed bf16 writes (cols 49..63 zeroed) ----
  {
    int h = tid >> 6, row = tid & 63;
    float*  ar = s_att + (h*NPAD + row)*AT_PITCH;
    __bf16* br = s_ab  + (h*NPAD + row)*AB_PITCH;
    if (row < NTOK) {
      float mx = -1e30f;
      for (int j = 0; j < NTOK; ++j) mx = fmaxf(mx, ar[j]);
      float sum = 0.0f;
      for (int j = 0; j < NTOK; ++j) { float e = __expf(ar[j] - mx); sum += e; ar[j] = e; }
      float inv = 1.0f / sum;
      #pragma unroll
      for (int g = 0; g < 8; ++g) {
        union { __bf16 h8[8]; uint4 q; } u;
        #pragma unroll
        for (int e = 0; e < 8; ++e) {
          int j = g*8 + e;
          u.h8[e] = (j < NTOK) ? (__bf16)(ar[j] * inv) : (__bf16)0.0f;
        }
        *(uint4*)(br + g*8) = u.q;
      }
    } else {
      uint4 z = make_uint4(0u, 0u, 0u, 0u);
      #pragma unroll
      for (int g = 0; g < 8; ++g) *(uint4*)(br + g*8) = z;
    }
  }
  __syncthreads();

  // ---- Phase 5: ctx = attn @ v ----
  for (int t = wave; t < 32; t += 8) {          // 4 heads x 4 M x 2 N tiles
    int h = t >> 3, mt = (t >> 1) & 3, nt3 = t & 1;
    v8f acc = {};
    const __bf16* arow = s_ab + (h*NPAD + mt*16 + l16) * AB_PITCH;
    const __bf16* brow = s_vT + (h*DH + nt3*16 + l16) * VT_PITCH;
    #pragma unroll
    for (int ks = 0; ks < 4; ++ks) {
      v16bf a = frag2(arow + ks*32 + hl*8, arow + ks*32 + 16 + hl*8);
      v16bf b = frag2(brow + ks*32 + hl*8, brow + ks*32 + 16 + hl*8);
      acc = __builtin_amdgcn_wmma_f32_16x16x32_bf16(false, a, false, b,
                                                    (short)0, acc, false, false);
    }
    int c = h*DH + nt3*16 + l16;
    __bf16* crow = s_ctx + (mt*16 + 8*hl)*X_PITCH + c;
    #pragma unroll
    for (int r = 0; r < 8; ++r)
      crow[r*X_PITCH] = (__bf16)acc[r];
  }
  __syncthreads();

  // ---- Phase 6: out = ctx @ proj_w + proj_b (f32 stores) ----
  float* outw = out + (size_t)w * NTOK * CH;
  for (int t = wave; t < 32; t += 8) {          // 4 M x 8 N tiles
    int mt = t >> 3, nt = t & 7;
    int c = nt*16 + l16;
    v8f acc = {};
    const __bf16* arow = s_ctx + (mt*16 + l16) * X_PITCH;
    const __bf16* brow = projT + (size_t)c * CH;
    #pragma unroll
    for (int ks = 0; ks < 4; ++ks) {
      v16bf a = frag2(arow + ks*32 + hl*8, arow + ks*32 + 16 + hl*8);
      v16bf b = frag2(brow + ks*32 + hl*8, brow + ks*32 + 16 + hl*8);
      acc = __builtin_amdgcn_wmma_f32_16x16x32_bf16(false, a, false, b,
                                                    (short)0, acc, false, false);
    }
    float pb = proj_b[c];
    int mbase = mt*16 + 8*hl;
    float* orow = outw + (size_t)mbase * CH + c;
    if (mt < 3) {                               // rows < 48: always in range
      #pragma unroll
      for (int r = 0; r < 8; ++r)
        orow[r*CH] = acc[r] + pb;
    } else {                                    // only row 48 is valid here
      if (hl == 0) orow[0] = acc[0] + pb;
    }
  }
}

extern "C" void kernel_launch(void* const* d_in, const int* in_sizes, int n_in,
                              void* d_out, int out_size, void* d_ws, size_t ws_size,
                              hipStream_t stream) {
  const float* x      = (const float*)d_in[0];
  const float* qkv_w  = (const float*)d_in[1];
  const float* qkv_b  = (const float*)d_in[2];
  const float* proj_w = (const float*)d_in[3];
  const float* proj_b = (const float*)d_in[4];
  const float* bias_t = (const float*)d_in[5];
  const float* mask   = (const float*)d_in[6];
  float* out = (float*)d_out;

  __bf16* wqkvT = (__bf16*)((char*)d_ws + WS_WQKVT);
  __bf16* projT = (__bf16*)((char*)d_ws + WS_PROJT);
  float*  rpb   = (float*) ((char*)d_ws + WS_RPB);

  int prep_total = C3*CH + CH*CH + NHEAD*NTOK*NTOK;
  swin_prep<<<(prep_total + 255)/256, 256, 0, stream>>>(
      qkv_w, proj_w, bias_t, wqkvT, projT, rpb);

  swin_win_attn<<<NWINDOWS, 256, SMEM_BYTES, stream>>>(
      x, qkv_b, proj_b, mask, wqkvT, projT, rpb, out);
}
